// COHF_32693291057571
// MI455X (gfx1250) — matrix operand order repaired and checked by reading, not compiled, not run on hardware
//
#include <hip/hip_runtime.h>

// Problem constants (match reference)
#define KN   4096
#define KF   512
#define KZ   256
#define KH   256
#define KE   64
#define KATT 128
#define KP   16
#define KB   128

typedef __attribute__((ext_vector_type(16))) __bf16 bf16x16;
typedef __attribute__((ext_vector_type(8)))  float  f32x8;
typedef unsigned short ushort_t;

__device__ __forceinline__ ushort_t f2bf(float f) {
  unsigned int u = __float_as_uint(f);
  u += 0x7FFFu + ((u >> 16) & 1u);          // round-to-nearest-even
  return (ushort_t)(u >> 16);
}

// ------------------------------------------------------------------
// Generic bf16 WMMA GEMM:
//   C[M,Nc] = act( row_scale[m] * (A @ B) + bias[n] )
// A: bf16 row-major [M,K] (lda), B: bf16 row-major [K,Nc] (ldb), C: f32 (ldc)
// Block = 256 threads (8 waves); block tile 128(M) x 128(N); K step 32.
// A tile is staged into LDS with the CDNA5 async global->LDS DMA
// (GLOBAL_LOAD_ASYNC_TO_LDS_B128, tracked by ASYNCcnt); the B tile needs a
// transpose-scatter so it uses regular load + ds_store. LDS layouts are
// arranged so WMMA fragments load as contiguous b128 reads matching the
// CDNA5 16-bit A (16x32) and B (32x16) per-lane layouts.
// act: 0 none, 1 relu, 2 tanh, 3 sigmoid(relu(x))
// ------------------------------------------------------------------
__global__ __launch_bounds__(256)
void gemm_bf16_kernel(const ushort_t* __restrict__ A, int lda,
                      const ushort_t* __restrict__ Bm, int ldb,
                      float* __restrict__ C, int ldc,
                      int M, int Nc, int K,
                      const float* __restrict__ row_scale,
                      const float* __restrict__ bias, int act)
{
  __shared__ __align__(16) ushort_t sA[128 * 32];   // [row][k] row-major
  __shared__ __align__(16) ushort_t sB[128 * 32];   // [ngrp][n%16][k]
  const int tid = threadIdx.x;
  const int w   = tid >> 5;       // wave 0..7  -> M strip
  const int L   = tid & 31;       // lane
  const int m0  = blockIdx.y * 128;
  const int n0  = blockIdx.x * 128;

  f32x8 acc[8];
  const f32x8 zero = {0.f,0.f,0.f,0.f,0.f,0.f,0.f,0.f};
  #pragma unroll
  for (int g = 0; g < 8; ++g) acc[g] = zero;

  // A fragment addressing (ISA 16-bit A 16x32 layout):
  //  lane L: M = L%16, e=0..7 -> K = e + (L>=16?8:0); e=8..15 -> K = 16+(e-8)+(L>=16?8:0)
  const int lrow = w * 16 + (L & 15);
  const int k0h  = (L >= 16) ? 8 : 0;
  // B fragment addressing (ISA 16-bit B 32x16 layout):
  //  lane L: N = L%16, e=0..7 -> K = e + (L>=16?8:0); e=8..15 -> K = 16+(e-8)+(L>=16?8:0)
  const int nl = L & 15;
  const int kb = (L < 16) ? 0 : 8;

  // per-thread staging coordinates (2 b128 chunks each for A and B)
  const int arow0 = tid >> 1,            acol0 = (tid & 1) << 3;       // chunks 0..255
  const int arow1 = (tid + 256) >> 1,    acol1 = ((tid + 256) & 1) << 3;
  // A chunk c: row = c>>2, col8 = (c&3)<<3  (reindexed: 512 chunks over 2 iters)

  for (int k0 = 0; k0 < K; k0 += 32) {
    // ---- stage A tile (128 x 32 bf16) via async DMA to LDS (ASYNCcnt)
    #pragma unroll
    for (int it = 0; it < 2; ++it) {
      int c = tid + it * 256;
      int row = c >> 2, col8 = (c & 3) << 3;
      unsigned ldsoff = (unsigned)(uintptr_t)(const void*)&sA[row * 32 + col8];
      const ushort_t* gp = &A[(size_t)(m0 + row) * lda + k0 + col8];
      asm volatile("global_load_async_to_lds_b128 %0, %1, off"
                   :: "v"(ldsoff),
                      "v"((unsigned long long)(uintptr_t)gp)
                   : "memory");
      if (k0 + 32 < K) __builtin_prefetch(gp + 32, 0, 1);
    }
    // ---- stage B tile transposed per 16-col group: sB[(n/16)*512 + (n%16)*32 + k]
    #pragma unroll
    for (int it = 0; it < 2; ++it) {
      int c = tid + it * 256;
      int k = c >> 4, n8 = (c & 15) << 3;
      uint4 val = make_uint4(0u, 0u, 0u, 0u);
      if (n0 + n8 < Nc) {
        const ushort_t* gp = &Bm[(size_t)(k0 + k) * ldb + n0 + n8];
        val = *(const uint4*)gp;
        if (k0 + 32 < K) __builtin_prefetch(gp + (size_t)32 * ldb, 0, 1);
      }
      const ushort_t* vs = (const ushort_t*)&val;
      #pragma unroll
      for (int j = 0; j < 8; ++j) {
        int n = n8 + j;
        sB[(n >> 4) * 512 + (n & 15) * 32 + k] = vs[j];
      }
    }
    // async DMA completion fence, then workgroup barrier
    asm volatile("s_wait_asynccnt 0x0" ::: "memory");
    __syncthreads();

    // ---- load all fragments first (distinct regs), then burst 8 WMMAs
    union { bf16x16 v; uint4 q[2]; } fa;
    fa.q[0] = *(const uint4*)&sA[lrow * 32 + k0h];
    fa.q[1] = *(const uint4*)&sA[lrow * 32 + 16 + k0h];
    union { bf16x16 v; uint4 q[2]; } fb[8];
    #pragma unroll
    for (int g = 0; g < 8; ++g) {
      const uint4* p = (const uint4*)&sB[g * 512 + nl * 32 + kb];
      fb[g].q[0] = p[0];
      fb[g].q[1] = p[2];
    }
    #pragma unroll
    for (int g = 0; g < 8; ++g) {
      acc[g] = __builtin_amdgcn_wmma_f32_16x16x32_bf16(
          false, fa.v, false, fb[g].v, (short)0, acc[g], false, false);
    }
    __syncthreads();
  }

  // ---- epilogue (ISA 32-bit C/D layout: r -> M = r + (L>=16?8:0), N = L%16)
  const int mh = (L >= 16) ? 8 : 0;
  #pragma unroll
  for (int g = 0; g < 8; ++g) {
    int n = n0 + g * 16 + nl;
    if (n >= Nc) continue;
    union { f32x8 v; float f[8]; } u; u.v = acc[g];
    #pragma unroll
    for (int r = 0; r < 8; ++r) {
      int m = m0 + w * 16 + mh + r;
      float v = u.f[r];
      if (row_scale) v *= row_scale[m];
      if (bias)      v += bias[n];
      if (act == 1)      v = fmaxf(v, 0.f);
      else if (act == 2) v = tanhf(v);
      else if (act == 3) { v = fmaxf(v, 0.f); v = 1.f / (1.f + __expf(-v)); }
      C[(size_t)m * ldc + n] = v;
    }
  }
  (void)arow0; (void)acol0; (void)arow1; (void)acol1;
}

// ---------------- elementwise / helper kernels ----------------

__global__ void f2bf_kernel(const float* __restrict__ s, ushort_t* __restrict__ d, size_t n) {
  size_t i = (size_t)blockIdx.x * blockDim.x + threadIdx.x;
  size_t st = (size_t)gridDim.x * blockDim.x;
  for (; i < n; i += st) d[i] = f2bf(s[i]);
}

// out = src with diagonal forced to 1   (N x N, N == KN)
__global__ void diag1_copy_kernel(const float* __restrict__ s, float* __restrict__ d) {
  size_t i = (size_t)blockIdx.x * blockDim.x + threadIdx.x;
  size_t st = (size_t)gridDim.x * blockDim.x;
  size_t total = (size_t)KN * KN;
  for (; i < total; i += st) {
    int n = (int)(i >> 12), m = (int)(i & (KN - 1));
    d[i] = (n == m) ? 1.f : s[i];
  }
}

// diff effective adjacency: d = diff + 0.5*(c0+c1), diag -> 1
__global__ void diff_adj_kernel(const float* __restrict__ dd, const float* __restrict__ c0,
                                const float* __restrict__ c1, float* __restrict__ out) {
  size_t i = (size_t)blockIdx.x * blockDim.x + threadIdx.x;
  size_t st = (size_t)gridDim.x * blockDim.x;
  size_t total = (size_t)KN * KN;
  for (; i < total; i += st) {
    int n = (int)(i >> 12), m = (int)(i & (KN - 1));
    out[i] = (n == m) ? 1.f : (dd[i] + 0.5f * (c0[i] + c1[i]));
  }
}

// F1 = w0*0.5*(c0+c1) + w1*0.5*(d0+d1)
__global__ void ewF1_kernel(const float* __restrict__ c0, const float* __restrict__ c1,
                            const float* __restrict__ d0, const float* __restrict__ d1,
                            const float* __restrict__ wb, float* __restrict__ out) {
  float w0 = wb[0], w1 = wb[1];
  size_t i = (size_t)blockIdx.x * blockDim.x + threadIdx.x;
  size_t st = (size_t)gridDim.x * blockDim.x;
  size_t total = (size_t)KN * KN;
  for (; i < total; i += st)
    out[i] = 0.5f * (w0 * (c0[i] + c1[i]) + w1 * (d0[i] + d1[i]));
}

// F2 = w0*relu(P1) + w1*relu(P2)
__global__ void ewF2_kernel(const float* __restrict__ p1, const float* __restrict__ p2,
                            const float* __restrict__ wb, float* __restrict__ out) {
  float w0 = wb[0], w1 = wb[1];
  size_t i = (size_t)blockIdx.x * blockDim.x + threadIdx.x;
  size_t st = (size_t)gridDim.x * blockDim.x;
  size_t total = (size_t)KN * KN;
  for (; i < total; i += st)
    out[i] = w0 * fmaxf(p1[i], 0.f) + w1 * fmaxf(p2[i], 0.f);
}

// dst = src + src^T, diag forced to 1. Tiled 32x32 via LDS (dst != src).
__global__ void sym_diag1_kernel(const float* __restrict__ src, float* __restrict__ dst) {
  __shared__ float tr[32][33];
  int n0 = blockIdx.y * 32, m0 = blockIdx.x * 32;
  for (int r = threadIdx.y; r < 32; r += 8)
    tr[r][threadIdx.x] = src[(size_t)(m0 + r) * KN + n0 + threadIdx.x];
  __syncthreads();
  for (int r = threadIdx.y; r < 32; r += 8) {
    int n = n0 + r, m = m0 + threadIdx.x;
    float v = src[(size_t)n * KN + m] + tr[threadIdx.x][r];
    dst[(size_t)n * KN + m] = (n == m) ? 1.f : v;
  }
}

// d[row] = rsqrt(max(rowsum, 1))
__global__ void rowsum_rsqrt_kernel(const float* __restrict__ A, float* __restrict__ d) {
  __shared__ float sh[256];
  int row = blockIdx.x, t = threadIdx.x;
  float s = 0.f;
  for (int m = t; m < KN; m += 256) s += A[(size_t)row * KN + m];
  sh[t] = s; __syncthreads();
  for (int st = 128; st > 0; st >>= 1) { if (t < st) sh[t] += sh[t + st]; __syncthreads(); }
  if (t == 0) d[row] = rsqrtf(fmaxf(sh[0], 1.f));
}

// dst_bf16[n, c] = bf16( d[n] * src[n, c] )
__global__ void rowscale_bf16_kernel(const float* __restrict__ src, const float* __restrict__ d,
                                     ushort_t* __restrict__ dst, int cols, size_t total) {
  size_t i = (size_t)blockIdx.x * blockDim.x + threadIdx.x;
  size_t st = (size_t)gridDim.x * blockDim.x;
  for (; i < total; i += st) {
    size_t n = i / cols;
    dst[i] = f2bf(d[n] * src[i]);
  }
}

// svec[row] = dot(T[row,:], w2)
__global__ void dot_w2_kernel(const float* __restrict__ T, const float* __restrict__ w2,
                              float* __restrict__ svec, int cols) {
  __shared__ float sh[128];
  int row = blockIdx.x, t = threadIdx.x;
  float s = 0.f;
  for (int c = t; c < cols; c += 128) s += T[(size_t)row * cols + c] * w2[c];
  sh[t] = s; __syncthreads();
  for (int st = 64; st > 0; st >>= 1) { if (t < st) sh[t] += sh[t + st]; __syncthreads(); }
  if (t == 0) svec[row] = sh[0];
}

__global__ void mean_reduce_kernel(const float* __restrict__ v, int n,
                                   float* __restrict__ out, int k) {
  __shared__ float sh[256];
  int t = threadIdx.x;
  float s = 0.f;
  for (int i = t; i < n; i += 256) s += v[i];
  sh[t] = s; __syncthreads();
  for (int st = 128; st > 0; st >>= 1) { if (t < st) sh[t] += sh[t + st]; __syncthreads(); }
  if (t == 0) out[k] = sh[0] / (float)n;
}

__global__ void softmax_beta_kernel(const float* __restrict__ mvec,
                                    float* __restrict__ beta, int Kk) {
  if (threadIdx.x == 0 && blockIdx.x == 0) {
    float mx = mvec[0];
    for (int i = 1; i < Kk; ++i) mx = fmaxf(mx, mvec[i]);
    float s = 0.f;
    for (int i = 0; i < Kk; ++i) s += __expf(mvec[i] - mx);
    for (int i = 0; i < Kk; ++i) beta[i] = __expf(mvec[i] - mx) / s;
  }
}

__global__ void weighted_sum_kernel(float* __restrict__ out,
                                    const float* __restrict__ r0, const float* __restrict__ r1,
                                    const float* __restrict__ r2, const float* __restrict__ r3,
                                    const float* __restrict__ beta, int Kk, size_t n) {
  size_t i = (size_t)blockIdx.x * blockDim.x + threadIdx.x;
  size_t st = (size_t)gridDim.x * blockDim.x;
  for (; i < n; i += st) {
    float v = beta[0] * r0[i] + beta[1] * r1[i];
    if (Kk > 2) v += beta[2] * r2[i] + beta[3] * r3[i];
    out[i] = v;
  }
}

// _adj_gen operand:  G[n, p*E+e] = (ctx[n,e]*Wt[p,e]) / (max(||ctx[n]*Wt[p]||,1e-12)*sqrt(P))
// Also writes Gt = G^T for the row-major WMMA GEMM.
__global__ void build_G_kernel(const float* __restrict__ ctx, const float* __restrict__ Wt,
                               ushort_t* __restrict__ G, ushort_t* __restrict__ Gt) {
  __shared__ float sh[64];
  int e = threadIdx.x;           // 0..63
  int idx = blockIdx.x;          // 0..N*P-1
  int n = idx >> 4, p = idx & 15;
  float v = ctx[(size_t)n * KE + e] * Wt[p * KE + e];
  sh[e] = v * v;
  __syncthreads();
  for (int s = 32; s > 0; s >>= 1) { if (e < s) sh[e] += sh[e + s]; __syncthreads(); }
  float norm = sqrtf(sh[0]);
  float sc = 1.f / (fmaxf(norm, 1e-12f) * 4.0f);   // sqrt(P)=4
  ushort_t b = f2bf(v * sc);
  G [(size_t)n * (KP * KE) + p * KE + e] = b;
  Gt[(size_t)(p * KE + e) * KN + n] = b;
}

__global__ void avg2_kernel(const float* __restrict__ a, const float* __restrict__ b,
                            float* __restrict__ o, size_t n) {
  size_t i = (size_t)blockIdx.x * blockDim.x + threadIdx.x;
  size_t st = (size_t)gridDim.x * blockDim.x;
  for (; i < n; i += st) o[i] = 0.5f * (a[i] + b[i]);
}

__global__ void gather_h_kernel(const float* __restrict__ z1, const float* __restrict__ z2,
                                const int* __restrict__ xs, float* __restrict__ out) {
  int i = blockIdx.x * blockDim.x + threadIdx.x;
  if (i >= KB * 2 * KZ) return;
  int b = i / (2 * KZ), j = i % (2 * KZ);
  int x = xs[b];
  out[i] = (j < KZ) ? z1[(size_t)x * KZ + j] : z2[(size_t)x * KZ + (j - KZ)];
}

// ------------------------------------------------------------------
extern "C" void kernel_launch(void* const* d_in, const int* in_sizes, int n_in,
                              void* d_out, int out_size, void* d_ws, size_t ws_size,
                              hipStream_t stream) {
  (void)in_sizes; (void)n_in; (void)out_size; (void)ws_size;

  const float* comm0   = (const float*)d_in[0];
  const float* comm1   = (const float*)d_in[1];
  const float* diff0   = (const float*)d_in[2];
  const float* diff1   = (const float*)d_in[3];
  const float* feats   = (const float*)d_in[4];
  const float* Wc_gnn  = (const float*)d_in[5];
  const float* bc_gnn  = (const float*)d_in[6];
  const float* Wd_gnn  = (const float*)d_in[7];
  const float* bd_gnn  = (const float*)d_in[8];
  const float* Wcatt1  = (const float*)d_in[9];
  const float* bcatt1  = (const float*)d_in[10];
  const float* wcatt2  = (const float*)d_in[11];
  const float* Wdatt1  = (const float*)d_in[12];
  const float* bdatt1  = (const float*)d_in[13];
  const float* wdatt2  = (const float*)d_in[14];
  const float* Wmu_c   = (const float*)d_in[15];
  const float* bmu_c   = (const float*)d_in[16];
  const float* Wvar_c  = (const float*)d_in[17];
  const float* bvar_c  = (const float*)d_in[18];
  const float* Wmu_d   = (const float*)d_in[19];
  const float* bmu_d   = (const float*)d_in[20];
  const float* Wvar_d  = (const float*)d_in[21];
  const float* bvar_d  = (const float*)d_in[22];
  const float* Wz1g1   = (const float*)d_in[23];
  const float* bz1g1   = (const float*)d_in[24];
  const float* Wz1g2   = (const float*)d_in[25];
  const float* bz1g2   = (const float*)d_in[26];
  const float* wbz1    = (const float*)d_in[27];
  const float* Wt_comm = (const float*)d_in[28];
  const float* Wt_diff = (const float*)d_in[29];
  const float* Wz2g1   = (const float*)d_in[30];
  const float* bz2g1   = (const float*)d_in[31];
  const float* Wz2g2   = (const float*)d_in[32];
  const float* bz2g2   = (const float*)d_in[33];
  const float* wbz2    = (const float*)d_in[34];
  const int*   xs      = (const int*)d_in[35];

  float* out       = (float*)d_out;
  float* out_mu_c  = out + (size_t)KB * 2 * KZ;
  float* out_var_c = out_mu_c + (size_t)KN * KE;
  float* out_mu_d  = out_var_c + (size_t)KN * KE;
  float* out_var_d = out_mu_d + (size_t)KN * KE;

  // ---- workspace bump allocator
  char* cur = (char*)d_ws;
  auto alloc = [&](size_t b) -> void* {
    void* p = (void*)cur; cur += (b + 255) & ~(size_t)255; return p;
  };
  const size_t NN = (size_t)KN * KN;
  float*    Af     = (float*)alloc(NN * 4);
  float*    P1     = (float*)alloc(NN * 4);
  float*    P2     = (float*)alloc(NN * 4);
  ushort_t* Abf    = (ushort_t*)alloc(NN * 2);
  ushort_t* featsb = (ushort_t*)alloc((size_t)KN * KF * 2);
  ushort_t* Wb_c   = (ushort_t*)alloc((size_t)KF * KH * 2);
  ushort_t* Wb_d   = (ushort_t*)alloc((size_t)KF * KH * 2);
  ushort_t* Wb_z1a = (ushort_t*)alloc((size_t)KF * KZ * 2);
  ushort_t* Wb_z2a = (ushort_t*)alloc((size_t)KF * KZ * 2);
  ushort_t* Wb_z1b = (ushort_t*)alloc((size_t)KZ * KZ * 2);
  ushort_t* Wb_z2b = (ushort_t*)alloc((size_t)KZ * KZ * 2);
  ushort_t* Wb_ca  = (ushort_t*)alloc((size_t)KH * KATT * 2);
  ushort_t* Wb_da  = (ushort_t*)alloc((size_t)KH * KATT * 2);
  ushort_t* Wb_mc  = (ushort_t*)alloc((size_t)KH * KE * 2);
  ushort_t* Wb_vc  = (ushort_t*)alloc((size_t)KH * KE * 2);
  ushort_t* Wb_md  = (ushort_t*)alloc((size_t)KH * KE * 2);
  ushort_t* Wb_vd  = (ushort_t*)alloc((size_t)KH * KE * 2);
  float*    XWc    = (float*)alloc((size_t)KN * KH * 4);
  float*    XWd    = (float*)alloc((size_t)KN * KH * 4);
  float*    XWz1   = (float*)alloc((size_t)KN * KZ * 4);
  float*    XWz2   = (float*)alloc((size_t)KN * KZ * 4);
  ushort_t* Yb     = (ushort_t*)alloc((size_t)KN * KH * 2);
  float*    rep0   = (float*)alloc((size_t)KN * KH * 4);
  float*    rep1   = (float*)alloc((size_t)KN * KH * 4);
  float*    rep2   = (float*)alloc((size_t)KN * KH * 4);
  float*    rep3   = (float*)alloc((size_t)KN * KH * 4);
  float*    Tbuf   = (float*)alloc((size_t)KN * KATT * 4);
  float*    svec   = (float*)alloc((size_t)KN * 4);
  float*    mvec   = (float*)alloc(64);
  float*    beta   = (float*)alloc(64);
  float*    agg_c  = (float*)alloc((size_t)KN * KH * 4);
  float*    agg_d  = (float*)alloc((size_t)KN * KH * 4);
  float*    dvec   = (float*)alloc((size_t)KN * 4);
  float*    U1     = (float*)alloc((size_t)KN * KZ * 4);
  float*    U2     = (float*)alloc((size_t)KN * KZ * 4);
  float*    Utmp   = (float*)alloc((size_t)KN * KZ * 4);
  float*    z1     = (float*)alloc((size_t)KN * KZ * 4);
  float*    z2     = (float*)alloc((size_t)KN * KZ * 4);
  ushort_t* G      = (ushort_t*)alloc((size_t)KN * KP * KE * 2);
  ushort_t* Gt     = (ushort_t*)alloc((size_t)KN * KP * KE * 2);

  const int EWB = 32768;  // elementwise grid blocks (grid-stride)

  auto cvt = [&](const float* s, ushort_t* d, size_t n) {
    int blk = (int)((n + 255) / 256); if (blk > EWB) blk = EWB;
    f2bf_kernel<<<blk, 256, 0, stream>>>(s, d, n);
  };
  auto gemm = [&](const ushort_t* A, int lda, const ushort_t* B, int ldb,
                  float* C, int ldc, int M, int Nc, int K,
                  const float* rs, const float* bias, int act) {
    dim3 g((Nc + 127) / 128, M / 128);
    gemm_bf16_kernel<<<g, 256, 0, stream>>>(A, lda, B, ldb, C, ldc, M, Nc, K, rs, bias, act);
  };

  // ---- stage 0: convert feats + all weights to bf16
  cvt(feats,  featsb, (size_t)KN * KF);
  cvt(Wc_gnn, Wb_c,  (size_t)KF * KH);   cvt(Wd_gnn, Wb_d,  (size_t)KF * KH);
  cvt(Wz1g1,  Wb_z1a,(size_t)KF * KZ);   cvt(Wz2g1,  Wb_z2a,(size_t)KF * KZ);
  cvt(Wz1g2,  Wb_z1b,(size_t)KZ * KZ);   cvt(Wz2g2,  Wb_z2b,(size_t)KZ * KZ);
  cvt(Wcatt1, Wb_ca, (size_t)KH * KATT); cvt(Wdatt1, Wb_da, (size_t)KH * KATT);
  cvt(Wmu_c,  Wb_mc, (size_t)KH * KE);   cvt(Wvar_c, Wb_vc, (size_t)KH * KE);
  cvt(Wmu_d,  Wb_md, (size_t)KH * KE);   cvt(Wvar_d, Wb_vd, (size_t)KH * KE);

  // ---- stage 1: XW = feats @ W   (M=4096, K=512, N=256)
  gemm(featsb, KF, Wb_c,  KH, XWc,  KH, KN, KH, KF, nullptr, nullptr, 0);
  gemm(featsb, KF, Wb_d,  KH, XWd,  KH, KN, KH, KF, nullptr, nullptr, 0);
  gemm(featsb, KF, Wb_z1a,KZ, XWz1, KZ, KN, KZ, KF, nullptr, nullptr, 0);
  gemm(featsb, KF, Wb_z2a,KZ, XWz2, KZ, KN, KZ, KF, nullptr, nullptr, 0);

  // shared GCN propagation step: Af holds effective adjacency (diag==1)
  auto gcn_prop = [&](const float* A_eff, const float* XW, const float* bias_v, float* rep) {
    rowsum_rsqrt_kernel<<<KN, 256, 0, stream>>>(A_eff, dvec);
    cvt(A_eff, Abf, NN);
    rowscale_bf16_kernel<<<EWB, 256, 0, stream>>>(XW, dvec, Yb, KH, (size_t)KN * KH);
    gemm(Abf, KN, Yb, KH, rep, KH, KN, KH, KN, dvec, bias_v, 1);
  };

  // ---- stage 2: comm reps
  diag1_copy_kernel<<<EWB, 256, 0, stream>>>(comm0, Af);
  gcn_prop(Af, XWc, bc_gnn, rep0);
  diag1_copy_kernel<<<EWB, 256, 0, stream>>>(comm1, Af);
  gcn_prop(Af, XWc, bc_gnn, rep1);
  // diff reps (adj = diff_i + comm_mean)
  diff_adj_kernel<<<EWB, 256, 0, stream>>>(diff0, comm0, comm1, Af);
  gcn_prop(Af, XWd, bd_gnn, rep2);
  diff_adj_kernel<<<EWB, 256, 0, stream>>>(diff1, comm0, comm1, Af);
  gcn_prop(Af, XWd, bd_gnn, rep3);

  // ---- stage 3: attention
  auto att_score = [&](const float* rep, const ushort_t* W1b, const float* b1,
                       const float* w2, int k) {
    cvt(rep, Yb, (size_t)KN * KH);
    gemm(Yb, KH, W1b, KATT, Tbuf, KATT, KN, KATT, KH, nullptr, b1, 2);  // tanh
    dot_w2_kernel<<<KN, 128, 0, stream>>>(Tbuf, w2, svec, KATT);
    mean_reduce_kernel<<<1, 256, 0, stream>>>(svec, KN, mvec, k);
  };
  // comm attention (K=2 over rep0, rep1)
  att_score(rep0, Wb_ca, bcatt1, wcatt2, 0);
  att_score(rep1, Wb_ca, bcatt1, wcatt2, 1);
  softmax_beta_kernel<<<1, 32, 0, stream>>>(mvec, beta, 2);
  weighted_sum_kernel<<<4096, 256, 0, stream>>>(agg_c, rep0, rep1, rep0, rep0,
                                                beta, 2, (size_t)KN * KH);
  // diff attention (K=4 over rep0..rep3)
  att_score(rep0, Wb_da, bdatt1, wdatt2, 0);
  att_score(rep1, Wb_da, bdatt1, wdatt2, 1);
  att_score(rep2, Wb_da, bdatt1, wdatt2, 2);
  att_score(rep3, Wb_da, bdatt1, wdatt2, 3);
  softmax_beta_kernel<<<1, 32, 0, stream>>>(mvec, beta, 4);
  weighted_sum_kernel<<<4096, 256, 0, stream>>>(agg_d, rep0, rep1, rep2, rep3,
                                                beta, 4, (size_t)KN * KH);

  // ---- stage 4: mu / var heads (write directly into d_out)
  cvt(agg_c, Yb, (size_t)KN * KH);
  gemm(Yb, KH, Wb_mc, KE, out_mu_c,  KE, KN, KE, KH, nullptr, bmu_c,  1);
  gemm(Yb, KH, Wb_vc, KE, out_var_c, KE, KN, KE, KH, nullptr, bvar_c, 3);
  cvt(agg_d, Yb, (size_t)KN * KH);
  gemm(Yb, KH, Wb_md, KE, out_mu_d,  KE, KN, KE, KH, nullptr, bmu_d,  1);
  gemm(Yb, KH, Wb_vd, KE, out_var_d, KE, KN, KE, KH, nullptr, bvar_d, 3);

  // shared 2-layer GCN z-path (A_eff in P1 with diag==1)
  auto z_path = [&](const float* XW, const ushort_t* W2b, const float* b1,
                    const float* b2, float* zout) {
    rowsum_rsqrt_kernel<<<KN, 256, 0, stream>>>(P1, dvec);
    cvt(P1, Abf, NN);
    rowscale_bf16_kernel<<<EWB, 256, 0, stream>>>(XW, dvec, Yb, KZ, (size_t)KN * KZ);
    gemm(Abf, KN, Yb, KZ, U1, KZ, KN, KZ, KN, dvec, b1, 1);            // U1 = relu(gcn1)
    cvt(U1, Yb, (size_t)KN * KZ);
    gemm(Yb, KZ, W2b, KZ, Utmp, KZ, KN, KZ, KZ, nullptr, nullptr, 0);  // U1 @ W2
    rowscale_bf16_kernel<<<EWB, 256, 0, stream>>>(Utmp, dvec, Yb, KZ, (size_t)KN * KZ);
    gemm(Abf, KN, Yb, KZ, U2, KZ, KN, KZ, KN, dvec, b2, 1);            // U2 = relu(gcn2)
    avg2_kernel<<<4096, 256, 0, stream>>>(U1, U2, zout, (size_t)KN * KZ);
  };

  // ---- stage 5: z1 (final_A1 = F1 + F1^T, diag 1)
  ewF1_kernel<<<EWB, 256, 0, stream>>>(comm0, comm1, diff0, diff1, wbz1, Af);
  {
    dim3 g(KN / 32, KN / 32);
    sym_diag1_kernel<<<g, dim3(32, 8), 0, stream>>>(Af, P1);
  }
  z_path(XWz1, Wb_z1b, bz1g1, bz1g2, z1);

  // ---- stage 6: _adj_gen Gram matrices: att = G @ G^T
  build_G_kernel<<<KN * KP, 64, 0, stream>>>(out_mu_c, Wt_comm, G, Gt);
  gemm(G, KP * KE, Gt, KN, P1, KN, KN, KN, KP * KE, nullptr, nullptr, 0);
  build_G_kernel<<<KN * KP, 64, 0, stream>>>(out_mu_d, Wt_diff, G, Gt);
  gemm(G, KP * KE, Gt, KN, P2, KN, KN, KN, KP * KE, nullptr, nullptr, 0);

  // ---- stage 7: z2 (final_A2 = F2 + F2^T, diag 1, F2 = w0*relu(ca)+w1*relu(da))
  ewF2_kernel<<<EWB, 256, 0, stream>>>(P1, P2, wbz2, Af);
  {
    dim3 g(KN / 32, KN / 32);
    sym_diag1_kernel<<<g, dim3(32, 8), 0, stream>>>(Af, P1);
  }
  z_path(XWz2, Wb_z2b, bz2g1, bz2g2, z2);

  // ---- stage 8: gather h = concat(z1, z2)[xs]
  gather_h_kernel<<<(KB * 2 * KZ + 255) / 256, 256, 0, stream>>>(z1, z2, xs, out);
}
